// HPGGATLayer_63402307223552
// MI455X (gfx1250) — compile-verified, matrix operand order checked by reading, not run on hardware
//
#include <hip/hip_runtime.h>
#include <math.h>

#define IN_FEATS  128
#define OUT_FEATS 16
#define NUM_HEADS 8
#define PROJ      (OUT_FEATS * NUM_HEADS)   // 128

typedef __attribute__((ext_vector_type(16))) _Float16 v16h;
typedef __attribute__((ext_vector_type(8)))  float    v8f;

__device__ __forceinline__ float lrelu(float x) { return x > 0.0f ? x : 0.2f * x; }

// IEEE-order float atomic max: signed-int max for v>=0, unsigned-int min for v<0.
__device__ __forceinline__ void atomic_max_float(float* addr, float value) {
  if (value >= 0.0f)
    atomicMax((int*)addr, __float_as_int(value));
  else
    atomicMin((unsigned int*)addr, __float_as_uint(value));
}

// ---------------------------------------------------------------------------
// K0: zero output, init segment max to -inf, segment sum to 0
// ---------------------------------------------------------------------------
__global__ void k_init(float* __restrict__ out, float* __restrict__ mx,
                       float* __restrict__ sm, int n) {
  int i = blockIdx.x * blockDim.x + threadIdx.x;
  if (i < n * OUT_FEATS) out[i] = 0.0f;
  if (i < n * NUM_HEADS) { mx[i] = -INFINITY; sm[i] = 0.0f; }
}

// ---------------------------------------------------------------------------
// K1: h_trans = h @ W_node via v_wmma_f32_16x16x32_f16.
// One wave computes one 16x16 output tile; K-loop 128 in steps of 32.
// A (16x32 f16): lanes 0-15 hold row M=lane, K = {0..7,16..23};
//                lanes 16-31 hold same rows, K shifted by +8.
// B (32x16 f16): lane holds column N=lane&15 with the same K pattern.
// C (16x16 f32): lane L holds N=L&15, VGPR i -> M = i + 8*(L>>4).
// ---------------------------------------------------------------------------
__global__ __launch_bounds__(128) void k_gemm(const float* __restrict__ H,
                                              const float* __restrict__ W,
                                              float* __restrict__ HT,
                                              int mtiles) {
  const int wave = threadIdx.x >> 5;
  const int lane = threadIdx.x & 31;
  const int tile = blockIdx.x * 4 + wave;
  if (tile >= mtiles * 8) return;            // wave-uniform; exact grid in practice
  const int mt = tile >> 3;
  const int nt = tile & 7;
  const int r  = lane & 15;
  const int kh = lane >> 4;                  // 0 or 1: K half select

  const float* arow = H + (size_t)(mt * 16 + r) * IN_FEATS;
  const float* bcol = W + nt * 16 + r;

  v8f c = {};
#pragma unroll
  for (int k0 = 0; k0 < IN_FEATS; k0 += 32) {
    v16h a, b;
#pragma unroll
    for (int i = 0; i < 16; ++i) {
      const int k = (i & 7) + 8 * kh + 16 * (i >> 3);
      a[i] = (_Float16)arow[k0 + k];
      b[i] = (_Float16)bcol[(size_t)(k0 + k) * PROJ];
    }
    // (neg_a, A, neg_b, B, c_mod, C, reuse_a, reuse_b)
    c = __builtin_amdgcn_wmma_f32_16x16x32_f16(false, a, false, b,
                                               (short)0, c, false, false);
  }
#pragma unroll
  for (int i = 0; i < 8; ++i)
    HT[(size_t)(mt * 16 + i + 8 * kh) * PROJ + nt * 16 + r] = c[i];
}

// ---------------------------------------------------------------------------
// K2: per-node attention scores  s_src[n,h] = <h_trans[n,h,:], a_src[h,:]>
// ---------------------------------------------------------------------------
__global__ void k_scores(const float* __restrict__ HT,
                         const float* __restrict__ a_src,
                         const float* __restrict__ a_dst,
                         float* __restrict__ ssrc, float* __restrict__ sdst,
                         int n) {
  int i = blockIdx.x * blockDim.x + threadIdx.x;
  if (i >= n * NUM_HEADS) return;
  const int node = i >> 3, hh = i & 7;
  const float* v = HT + (size_t)node * PROJ + hh * OUT_FEATS;
  float s0 = 0.0f, s1 = 0.0f;
#pragma unroll
  for (int f = 0; f < OUT_FEATS; ++f) {
    s0 += v[f] * a_src[hh * OUT_FEATS + f];
    s1 += v[f] * a_dst[hh * OUT_FEATS + f];
  }
  ssrc[i] = s0;
  sdst[i] = s1;
}

// ---------------------------------------------------------------------------
// K3: per-(edge,head) attention logit + segment atomic max over dst
// ---------------------------------------------------------------------------
__global__ void k_edge_max(const int* __restrict__ src, const int* __restrict__ dst,
                           const float* __restrict__ ef, const float* __restrict__ We,
                           const float* __restrict__ ssrc, const float* __restrict__ sdst,
                           float* __restrict__ mx, int E) {
  int i = blockIdx.x * blockDim.x + threadIdx.x;
  if (i >= E * NUM_HEADS) return;
  const int e = i >> 3, hh = i & 7;
  const int s = src[e], d = dst[e];
  const float a = lrelu(ssrc[s * NUM_HEADS + hh] + sdst[d * NUM_HEADS + hh] +
                        ef[e] * We[hh]);
  atomic_max_float(&mx[d * NUM_HEADS + hh], a);
}

// ---------------------------------------------------------------------------
// K4: exp(attn - max[dst]) + segment atomic sum over dst (attn recomputed)
// ---------------------------------------------------------------------------
__global__ void k_edge_sum(const int* __restrict__ src, const int* __restrict__ dst,
                           const float* __restrict__ ef, const float* __restrict__ We,
                           const float* __restrict__ ssrc, const float* __restrict__ sdst,
                           const float* __restrict__ mx, float* __restrict__ sm, int E) {
  int i = blockIdx.x * blockDim.x + threadIdx.x;
  if (i >= E * NUM_HEADS) return;
  const int e = i >> 3, hh = i & 7;
  const int s = src[e], d = dst[e];
  const float a = lrelu(ssrc[s * NUM_HEADS + hh] + sdst[d * NUM_HEADS + hh] +
                        ef[e] * We[hh]);
  const float ex = __expf(a - mx[d * NUM_HEADS + hh]);
  atomicAdd(&sm[d * NUM_HEADS + hh], ex);
}

// ---------------------------------------------------------------------------
// K5: message scatter, head-mean folded in.
// 16 lanes per edge (lane = output feature); per lane reduce over 8 heads,
// then a single atomicAdd of (1/8)*sum into out[dst, f]. h_trans row reads
// are contiguous 64B per head and L2-resident (25.6 MB << 192 MB L2).
// ---------------------------------------------------------------------------
__global__ __launch_bounds__(256) void k_edge_scatter(
    const int* __restrict__ src, const int* __restrict__ dst,
    const float* __restrict__ ef, const float* __restrict__ We,
    const float* __restrict__ ssrc, const float* __restrict__ sdst,
    const float* __restrict__ mx, const float* __restrict__ sm,
    const float* __restrict__ HT, float* __restrict__ out, int E) {
  const int e = blockIdx.x * 16 + (threadIdx.x >> 4);
  const int f = threadIdx.x & 15;
  if (e >= E) return;
  const int s = src[e], d = dst[e];
  const float efe = ef[e];
  float acc = 0.0f;
#pragma unroll
  for (int hh = 0; hh < NUM_HEADS; ++hh) {
    const float a  = lrelu(ssrc[s * NUM_HEADS + hh] + sdst[d * NUM_HEADS + hh] +
                           efe * We[hh]);
    const float al = __expf(a - mx[d * NUM_HEADS + hh]) /
                     fmaxf(sm[d * NUM_HEADS + hh], 1e-12f);
    acc += al * HT[(size_t)s * PROJ + hh * OUT_FEATS + f];
  }
  atomicAdd(&out[d * OUT_FEATS + f], acc * (1.0f / NUM_HEADS));
}

// ---------------------------------------------------------------------------
extern "C" void kernel_launch(void* const* d_in, const int* in_sizes, int n_in,
                              void* d_out, int out_size, void* d_ws, size_t ws_size,
                              hipStream_t stream) {
  const float* H  = (const float*)d_in[0];   // [N,128]
  const int*   EI = (const int*)d_in[1];     // [2,E]
  const float* EF = (const float*)d_in[2];   // [E,1]
  const float* Wn = (const float*)d_in[3];   // [128,128]
  const float* We = (const float*)d_in[4];   // [1,8]
  const float* As = (const float*)d_in[5];   // [8,16]
  const float* Ad = (const float*)d_in[6];   // [8,16]
  float* out = (float*)d_out;                // [N,16]

  const int n = in_sizes[0] / IN_FEATS;      // 50000
  const int E = in_sizes[1] / 2;             // 800000
  const int* src = EI;
  const int* dst = EI + E;

  // workspace: 32 MB total
  float* HT   = (float*)d_ws;                // n*128
  float* ssrc = HT   + (size_t)n * PROJ;     // n*8
  float* sdst = ssrc + (size_t)n * NUM_HEADS;
  float* mx   = sdst + (size_t)n * NUM_HEADS;
  float* sm   = mx   + (size_t)n * NUM_HEADS;

  k_init<<<(n * OUT_FEATS + 255) / 256, 256, 0, stream>>>(out, mx, sm, n);

  const int mtiles = (n + 15) / 16;          // 3125 (exact)
  const int tiles  = mtiles * 8;             // 25000
  k_gemm<<<(tiles + 3) / 4, 128, 0, stream>>>(H, Wn, HT, mtiles);

  k_scores<<<(n * NUM_HEADS + 255) / 256, 256, 0, stream>>>(HT, As, Ad, ssrc, sdst, n);

  const int eh = E * NUM_HEADS;
  k_edge_max<<<(eh + 255) / 256, 256, 0, stream>>>(src, dst, EF, We, ssrc, sdst, mx, E);
  k_edge_sum<<<(eh + 255) / 256, 256, 0, stream>>>(src, dst, EF, We, ssrc, sdst, mx, sm, E);
  k_edge_scatter<<<(E + 15) / 16, 256, 0, stream>>>(src, dst, EF, We, ssrc, sdst,
                                                    mx, sm, HT, out, E);
}
